// MaskGIT_VectorQuantizer_10986526343951
// MI455X (gfx1250) — matrix operand order repaired and checked by reading, not compiled, not run on hardware
//
#include <hip/hip_runtime.h>

typedef _Float16 v8h  __attribute__((ext_vector_type(8)));
typedef _Float16 v16h __attribute__((ext_vector_type(16)));
typedef float    v8f  __attribute__((ext_vector_type(8)));

#define T_TOK 32768   // 32 * 32 * 32 tokens
#define CDIM  256
#define KCODE 1024
#define HW    1024    // 32*32 spatial per image
#define NIMG  32

// ---------------------------------------------------------------------------
// Kernel 0: L2-normalize codebook rows -> f16 "en" (row-major K x C = B^T)
// ---------------------------------------------------------------------------
__global__ void vq_norm_emb(const float* __restrict__ emb, _Float16* __restrict__ en) {
    const int row  = blockIdx.x;        // 0..1023
    const int tid  = threadIdx.x;       // 0..255 == C
    const int lane = tid & 31, w = tid >> 5;
    float v = emb[row * CDIM + tid];
    float s = v * v;
    #pragma unroll
    for (int off = 16; off; off >>= 1) s += __shfl_xor(s, off, 32);
    __shared__ float wsum[8];
    if (lane == 0) wsum[w] = s;
    __syncthreads();
    float tot = 0.f;
    #pragma unroll
    for (int i = 0; i < 8; ++i) tot += wsum[i];
    const float scale = 1.0f / fmaxf(sqrtf(tot), 1e-12f);
    en[row * CDIM + tid] = (_Float16)(v * scale);
}

// ---------------------------------------------------------------------------
// Kernel 1: L2-normalize z tokens + transpose NCHW -> (T, C) f16 row-major.
// Block = 256 threads handles 32 tokens x 256 dims. Reads coalesced over
// spatial, writes coalesced b128 via an LDS transpose tile.
// ---------------------------------------------------------------------------
__global__ void vq_norm_z(const float* __restrict__ z, _Float16* __restrict__ zn) {
    const int tid   = threadIdx.x;
    const int x     = tid & 31;          // token within tile
    const int y     = tid >> 5;          // 0..7 (dim group)
    const int tBase = blockIdx.x * 32;   // global token base (tiles never cross n)
    const int n     = tBase >> 10;
    const int sBase = tBase & 1023;
    const float* zb = z + (size_t)n * CDIM * HW + sBase;  // (c, s) at zb[c*HW + x]

    float vals[32];
    float ss = 0.f;
    #pragma unroll
    for (int i = 0; i < 32; ++i) {
        const int c = y + (i << 3);
        const float v = zb[(size_t)c * HW + x];
        vals[i] = v; ss += v * v;
    }
    __shared__ float    psum[8][32];
    __shared__ float    scaleArr[32];
    __shared__ _Float16 tile[32][CDIM + 8];
    psum[y][x] = ss;
    __syncthreads();
    if (y == 0) {
        float t = 0.f;
        #pragma unroll
        for (int i = 0; i < 8; ++i) t += psum[i][x];
        scaleArr[x] = 1.0f / fmaxf(sqrtf(t), 1e-12f);
    }
    __syncthreads();
    const float sc = scaleArr[x];
    #pragma unroll
    for (int i = 0; i < 32; ++i) tile[x][y + (i << 3)] = (_Float16)(vals[i] * sc);
    __syncthreads();
    // cooperative coalesced copy out: (token, c) row-major
    const int r  = tid >> 3;             // token row 0..31
    const int ch = tid & 7;              // 32-dim chunk
    const _Float16* src = &tile[r][ch * 32];
    _Float16*       dst = zn + (size_t)(tBase + r) * CDIM + ch * 32;
    #pragma unroll
    for (int i = 0; i < 4; ++i)
        *(v8h*)(dst + i * 8) = *(const v8h*)(src + i * 8);
}

// ---------------------------------------------------------------------------
// Kernel 2: fused WMMA GEMM + argmax. 8 waves/block, each wave owns a
// 16-token M-tile with its A panel register-resident. The 16-code x 256-dim
// codebook tile (8 KB) is shared by all 8 waves: staged into a double-
// buffered LDS tile with global_load_async_to_lds_b128 (ASYNCcnt) so the
// next tile streams in while the current one feeds v_wmma from LDS.
// ---------------------------------------------------------------------------
__global__ void vq_wmma_argmax(const _Float16* __restrict__ zn,
                               const _Float16* __restrict__ en,
                               int* __restrict__ idxOut) {
    __shared__ _Float16 btile[2][16][CDIM];   // 2 x 8 KB double buffer

    const int tid   = threadIdx.x;
    const int lane  = tid & 31;
    const int wave  = tid >> 5;
    const int mBase = (blockIdx.x * 8 + wave) * 16;
    const int hi    = (lane < 16) ? 0 : 1;   // lane half
    const int lm    = lane & 15;

    // ---- async-copy plumbing: each thread moves 2 x 16B per 8 KB tile ----
    const unsigned myOff    = (unsigned)tid * 16u;
    const unsigned ldsBase0 = (unsigned)(uintptr_t)(&btile[0][0][0]) + myOff;
    const unsigned ldsBase1 = (unsigned)(uintptr_t)(&btile[1][0][0]) + myOff;
    const unsigned long long gBase =
        (unsigned long long)(uintptr_t)en + (unsigned long long)myOff;

    #define VQ_ISSUE_TILE(nT, ldsB)                                           \
        do {                                                                  \
            unsigned long long g0 = gBase + (unsigned long long)(nT) * 8192u; \
            unsigned l0 = (ldsB);                                             \
            asm volatile("global_load_async_to_lds_b128 %0, %1, off"          \
                         :: "v"(l0), "v"(g0) : "memory");                     \
            unsigned long long g1 = g0 + 4096u;                               \
            unsigned l1 = l0 + 4096u;                                         \
            asm volatile("global_load_async_to_lds_b128 %0, %1, off"          \
                         :: "v"(l1), "v"(g1) : "memory");                     \
        } while (0)

    // ---- preload A: 16 tokens x 256 dims, contiguous k = 16*half + s ----
    const _Float16* Arow  = zn + (size_t)(mBase + lm) * CDIM;
    const int       akoff = hi ? 16 : 0;
    union { v16h v; struct { v8h lo; v8h hi; } p; } a[8];
    #pragma unroll
    for (int k = 0; k < 8; ++k) {
        const _Float16* ap = Arow + k * 32 + akoff;
        a[k].p.lo = *(const v8h*)(ap);
        a[k].p.hi = *(const v8h*)(ap + 8);
    }

    const int bkoff = hi ? 16 : 0;
    float best[8];
    int   bidx[8];
    #pragma unroll
    for (int v = 0; v < 8; ++v) { best[v] = -3.0e38f; bidx[v] = 0; }

    VQ_ISSUE_TILE(0, ldsBase0);                       // prime the pipeline

    for (int t = 0; t < KCODE / 16; ++t) {
        if (t < 63) {
            VQ_ISSUE_TILE(t + 1, (t & 1) ? ldsBase0 : ldsBase1);
            // in-order completion: <=2 outstanding means tile t has landed
            asm volatile("s_wait_asynccnt 0x2" ::: "memory");
        } else {
            asm volatile("s_wait_asynccnt 0x0" ::: "memory");
        }
        __syncthreads();                              // tile t visible to block

        const _Float16* Brow = &btile[t & 1][lm][bkoff];
        v8f c = {};
        #pragma unroll
        for (int k = 0; k < 8; ++k) {
            const v16h b = *(const v16h*)(Brow + k * 32);   // ds_load_b128 x2
            c = __builtin_amdgcn_wmma_f32_16x16x32_f16(
                    false, a[k].v, false, b, (short)0, c, false, false);
        }
        const int cand = t * 16 + lm;
        #pragma unroll
        for (int v = 0; v < 8; ++v)
            if (c[v] > best[v]) { best[v] = c[v]; bidx[v] = cand; }

        __syncthreads();                              // done reading buf[t&1]
    }
    #undef VQ_ISSUE_TILE

    // ---- cross-lane argmax within each 16-lane half (N dimension) ----
    #pragma unroll
    for (int off = 8; off >= 1; off >>= 1) {
        #pragma unroll
        for (int v = 0; v < 8; ++v) {
            const float ov = __shfl_xor(best[v], off, 16);
            const int   oi = __shfl_xor(bidx[v], off, 16);
            if (ov > best[v] || (ov == best[v] && oi < bidx[v])) {
                best[v] = ov; bidx[v] = oi;
            }
        }
    }
    // lane 0 of each half owns rows M = v + 8*hi
    if (lm == 0) {
        #pragma unroll
        for (int v = 0; v < 8; ++v) idxOut[mBase + hi * 8 + v] = bidx[v];
    }
}

// ---------------------------------------------------------------------------
// Kernel 3: gather RAW codebook rows into NCHW output (coalesced writes;
// idx + emb reads hit L2).
// ---------------------------------------------------------------------------
__global__ void vq_gather(const float* __restrict__ emb,
                          const int* __restrict__ idx,
                          float* __restrict__ out) {
    const int t = blockIdx.x * blockDim.x + threadIdx.x;  // < 8388608
    const int s = t & 1023;
    const int c = (t >> 10) & 255;
    const int n = t >> 18;
    const int id = idx[(n << 10) + s];
    out[t] = emb[id * CDIM + c];
}

// ---------------------------------------------------------------------------
extern "C" void kernel_launch(void* const* d_in, const int* in_sizes, int n_in,
                              void* d_out, int out_size, void* d_ws, size_t ws_size,
                              hipStream_t stream) {
    const float* z   = (const float*)d_in[0];   // (32,256,32,32) fp32
    const float* emb = (const float*)d_in[1];   // (1024,256) fp32

    float* quant  = (float*)d_out;                             // 8,388,608 floats NCHW
    int*   idxOut = (int*)(quant + (size_t)NIMG * CDIM * HW);  // 32,768 int32

    _Float16* zn = (_Float16*)d_ws;                                     // 16 MB
    _Float16* en = (_Float16*)((char*)d_ws + (size_t)T_TOK * CDIM * 2); // 512 KB

    vq_norm_emb   <<<KCODE,            256, 0, stream>>>(emb, en);
    vq_norm_z     <<<T_TOK / 32,       256, 0, stream>>>(z, zn);
    vq_wmma_argmax<<<T_TOK / (16 * 8), 256, 0, stream>>>(zn, en, idxOut);
    vq_gather     <<<(NIMG * CDIM * HW) / 256, 256, 0, stream>>>(emb, idxOut, quant);
}
